// BBoxHead_52905407152449
// MI455X (gfx1250) — matrix operand order repaired and checked by reading, not compiled, not run on hardware
//
#include <hip/hip_runtime.h>
#include <hip/hip_bf16.h>
#include <math.h>

// ---------------------------------------------------------------------------
// Mask R-CNN box head on MI455X (gfx1250, wave32, WMMA).
// GEMM1 dominates (51.4 GFLOP, ~340 FLOP/byte) -> compute bound -> f16 WMMA
// with fp32 accumulation (V_WMMA_F32_16X16X32_F16). f16 A+W fit in 192MB L2.
// Async global->LDS (ASYNCcnt) double buffering feeds the WMMA pipeline.
// ---------------------------------------------------------------------------

typedef __attribute__((ext_vector_type(16))) _Float16 v16h;
typedef __attribute__((ext_vector_type(8)))  _Float16 h8;
typedef __attribute__((ext_vector_type(8)))  float    v8f;
typedef int vs4i __attribute__((vector_size(16)));   // matches builtin param

#define M_ROWS 2000
#define K_1    12544
#define H_DIM  1024
#define N_CLS  81
#define N_OUT3 405      // 81 logits + 324 deltas fused

#if __has_builtin(__builtin_amdgcn_global_load_async_to_lds_b128) && \
    __has_builtin(__builtin_amdgcn_s_wait_asynccnt)
#define USE_ASYNC_LDS 1
#else
#define USE_ASYNC_LDS 0
#endif

#define AS1 __attribute__((address_space(1)))
#define AS3 __attribute__((address_space(3)))

static __device__ __forceinline__ int imin_d(int a, int b) { return a < b ? a : b; }

// -------------------- elementwise f32 -> f16 cast --------------------------
__global__ void cast_f16_kernel(const float* __restrict__ src,
                                _Float16* __restrict__ dst, long n) {
  long i = (long)blockIdx.x * blockDim.x + threadIdx.x;
  long stride = (long)gridDim.x * blockDim.x;
  for (; i < n; i += stride) dst[i] = (_Float16)src[i];
}

// ---- transpose-cast: src[K][N] f32 (row-major) -> dst[(n+rowOff)][K] f16 ---
__global__ void tcast_kernel(const float* __restrict__ src,
                             _Float16* __restrict__ dst,
                             int K, int N, int rowOff, int dstStride) {
  long i = (long)blockIdx.x * blockDim.x + threadIdx.x;
  long total = (long)K * N;
  if (i >= total) return;
  int k = (int)(i / N);
  int c = (int)(i % N);
  dst[(long)(c + rowOff) * dstStride + k] = (_Float16)src[i];
}

// -------------------- concat logits_b | delta_b ----------------------------
__global__ void bias3_kernel(const float* __restrict__ lb,
                             const float* __restrict__ db,
                             float* __restrict__ bias3) {
  int i = blockIdx.x * blockDim.x + threadIdx.x;
  if (i < N_CLS) bias3[i] = lb[i];
  else if (i < N_OUT3) bias3[i] = db[i - N_CLS];
}

// ---------------------------------------------------------------------------
// WMMA GEMM: C[M][N] = A[M][K](f16) * Bt[N][K](f16, i.e. B col-major) + bias
// Block: 256 threads (8 waves). Block tile 128x128, K-step 32.
// Wave (wm in 0..3, wn in 0..1) owns a 32x64 subtile: 2 (M) x 4 (N) WMMA frags.
// Double-buffered LDS fed by GLOBAL_LOAD_ASYNC_TO_LDS_B128 when available.
// All 6 fragments are materialized before the 8 WMMAs so ds_load waits
// stagger instead of draining DScnt to 0 between WMMA pairs.
// ---------------------------------------------------------------------------
__global__ __launch_bounds__(256, 2)
void wmma_gemm_kernel(const _Float16* __restrict__ A,
                      const _Float16* __restrict__ Bt,
                      const float* __restrict__ bias,
                      float* __restrict__ C,
                      int M, int N, int K) {
  __shared__ __align__(16) _Float16 As[2][128][32];
  __shared__ __align__(16) _Float16 Bs[2][128][32];

  const int tid     = threadIdx.x;
  const int wave    = tid >> 5;
  const int lane    = tid & 31;
  const int halfsel = lane >> 4;   // 0: lanes 0-15, 1: lanes 16-31
  const int l15     = lane & 15;
  const int wm      = wave >> 1;   // 0..3 -> rows wm*32
  const int wn      = wave & 1;    // 0..1 -> cols wn*64
  const int m0      = blockIdx.x * 128;
  const int n0      = blockIdx.y * 128;

  // cooperative tile load mapping: 2 threads per row, 16 f16 each (2x16B)
  const int ldr = tid >> 1;            // 0..127
  const int ldk = (tid & 1) << 4;      // 0 or 16

  const long arow = (long)imin_d(m0 + ldr, M - 1) * K;
  const long brow = (long)imin_d(n0 + ldr, N - 1) * K;

  const v8f vzero = {0.f, 0.f, 0.f, 0.f, 0.f, 0.f, 0.f, 0.f};
  v8f acc[2][4];
#pragma unroll
  for (int s = 0; s < 2; ++s)
#pragma unroll
    for (int t = 0; t < 4; ++t) acc[s][t] = vzero;

  // issue one K-step tile (A:128x32, B:128x32) into LDS buffer `buf`
  auto issue_tile = [&](int buf, int k0) {
    const _Float16* ap = A + arow + k0 + ldk;
    const _Float16* bp = Bt + brow + k0 + ldk;
#if USE_ASYNC_LDS
    __builtin_amdgcn_global_load_async_to_lds_b128(
        (AS1 vs4i*)ap, (AS3 vs4i*)&As[buf][ldr][ldk], 0, 0);
    __builtin_amdgcn_global_load_async_to_lds_b128(
        (AS1 vs4i*)(ap + 8), (AS3 vs4i*)&As[buf][ldr][ldk + 8], 0, 0);
    __builtin_amdgcn_global_load_async_to_lds_b128(
        (AS1 vs4i*)bp, (AS3 vs4i*)&Bs[buf][ldr][ldk], 0, 0);
    __builtin_amdgcn_global_load_async_to_lds_b128(
        (AS1 vs4i*)(bp + 8), (AS3 vs4i*)&Bs[buf][ldr][ldk + 8], 0, 0);
#else
    *(uint4*)&As[buf][ldr][ldk]     = *(const uint4*)(ap);
    *(uint4*)&As[buf][ldr][ldk + 8] = *(const uint4*)(ap + 8);
    *(uint4*)&Bs[buf][ldr][ldk]     = *(const uint4*)(bp);
    *(uint4*)&Bs[buf][ldr][ldk + 8] = *(const uint4*)(bp + 8);
#endif
  };

  const int nsteps = K >> 5;          // K is a multiple of 32
  issue_tile(0, 0);

  for (int step = 0; step < nsteps; ++step) {
    const int buf = step & 1;
    const bool more = (step + 1) < nsteps;
    if (more) issue_tile(buf ^ 1, (step + 1) << 5);   // prefetch next tile

#if USE_ASYNC_LDS
    // wait until only the just-issued (next-tile) 4 async ops are in flight
    if (more) __builtin_amdgcn_s_wait_asynccnt(4);
    else      __builtin_amdgcn_s_wait_asynccnt(0);
#endif
    __syncthreads();    // all waves' data for `buf` now visible in LDS

    // A fragments (ISA 16-bit A 16x32 layout):
    // lane<16 : K {0..7} then {16..23}; lane>=16 : K {8..15} then {24..31}
    v16h afrag[2];
#pragma unroll
    for (int s = 0; s < 2; ++s) {
      const int row = wm * 32 + s * 16 + l15;
      h8 lo = *(const h8*)&As[buf][row][halfsel * 8];
      h8 hi = *(const h8*)&As[buf][row][16 + halfsel * 8];
      afrag[s] = __builtin_shufflevector(lo, hi, 0, 1, 2, 3, 4, 5, 6, 7,
                                         8, 9, 10, 11, 12, 13, 14, 15);
    }

    // B fragments (32x16): same K order as A: lane<16 K {0..7},{16..23}; lane>=16 K {8..15},{24..31}.
    // Load all four into distinct registers before any WMMA so the DS waits
    // stagger rather than serializing each WMMA pair on dscnt==0.
    v16h bfrag[4];
#pragma unroll
    for (int t = 0; t < 4; ++t) {
      const int nrow = wn * 64 + t * 16 + l15;
      h8 b0 = *(const h8*)&Bs[buf][nrow][halfsel * 8];
      h8 b1 = *(const h8*)&Bs[buf][nrow][16 + halfsel * 8];
      bfrag[t] = __builtin_shufflevector(b0, b1, 0, 1, 2, 3, 4, 5, 6, 7,
                                         8, 9, 10, 11, 12, 13, 14, 15);
    }

#pragma unroll
    for (int t = 0; t < 4; ++t) {
#pragma unroll
      for (int s = 0; s < 2; ++s) {
        acc[s][t] = __builtin_amdgcn_wmma_f32_16x16x32_f16(
            false, afrag[s], false, bfrag[t], (short)0, acc[s][t], false, false);
      }
    }
    __syncthreads();    // all waves done reading `buf` before it is refilled
  }

  // store: D VGPR r -> row r (lanes 0-15) / row 8+r (lanes 16-31), col = l15
#pragma unroll
  for (int s = 0; s < 2; ++s) {
#pragma unroll
    for (int t = 0; t < 4; ++t) {
      const int gcol = n0 + wn * 64 + t * 16 + l15;
#pragma unroll
      for (int r = 0; r < 8; ++r) {
        const int grow = m0 + wm * 32 + s * 16 + halfsel * 8 + r;
        if (grow < M && gcol < N)
          C[(long)grow * N + gcol] = acc[s][t][r] + bias[gcol];
      }
    }
  }
}

// ------------- BN (training) per-column stats -> scale/shift ---------------
__global__ void bn_stats_kernel(const float* __restrict__ X,
                                const float* __restrict__ gamma,
                                const float* __restrict__ beta,
                                float* __restrict__ scale,
                                float* __restrict__ shift,
                                int Mrows, int Ncols) {
  __shared__ float s1[256];
  __shared__ float s2[256];
  const int col = blockIdx.x;
  const int tid = threadIdx.x;
  float a = 0.f, b = 0.f;
  for (int r = tid; r < Mrows; r += 256) {
    float v = X[(long)r * Ncols + col];
    a += v; b += v * v;
  }
  s1[tid] = a; s2[tid] = b;
  __syncthreads();
  for (int off = 128; off > 0; off >>= 1) {
    if (tid < off) { s1[tid] += s1[tid + off]; s2[tid] += s2[tid + off]; }
    __syncthreads();
  }
  if (tid == 0) {
    float mean = s1[0] / (float)Mrows;
    float var  = s2[0] / (float)Mrows - mean * mean;   // biased, like jnp.var
    float sc   = gamma[col] * rsqrtf(var + 1e-3f);     // keras eps
    scale[col] = sc;
    shift[col] = beta[col] - mean * sc;
  }
}

// ------------- apply BN + ReLU, cast to f16 for next GEMM ------------------
__global__ void bn_relu_cast_kernel(const float* __restrict__ X,
                                    const float* __restrict__ scale,
                                    const float* __restrict__ shift,
                                    _Float16* __restrict__ Y,
                                    int Ncols, long total) {
  long i = (long)blockIdx.x * blockDim.x + threadIdx.x;
  long stride = (long)gridDim.x * blockDim.x;
  for (; i < total; i += stride) {
    int col = (int)(i % Ncols);
    float v = fmaf(X[i], scale[col], shift[col]);
    v = v > 0.f ? v : 0.f;
    Y[i] = (_Float16)v;
  }
}

// ------- per-row: emit logits, softmax probs, copy deltas into d_out -------
__global__ void head_out_kernel(const float* __restrict__ C3,
                                float* __restrict__ out) {
  const int row = blockIdx.x;       // 0..1999
  const int tid = threadIdx.x;      // 128 threads
  __shared__ float red[128];
  const float* crow = C3 + (long)row * N_OUT3;

  float v = (tid < N_CLS) ? crow[tid] : -3.0e38f;
  if (tid < N_CLS) out[(long)row * N_CLS + tid] = v;          // logits

  red[tid] = v;
  __syncthreads();
  for (int off = 64; off > 0; off >>= 1) {
    if (tid < off) red[tid] = fmaxf(red[tid], red[tid + off]);
    __syncthreads();
  }
  float m = red[0];
  __syncthreads();

  float e = (tid < N_CLS) ? expf(v - m) : 0.f;
  red[tid] = e;
  __syncthreads();
  for (int off = 64; off > 0; off >>= 1) {
    if (tid < off) red[tid] += red[tid + off];
    __syncthreads();
  }
  float inv = 1.0f / red[0];
  if (tid < N_CLS)
    out[(long)M_ROWS * N_CLS + (long)row * N_CLS + tid] = e * inv;  // probs

  const long dbase = 2L * M_ROWS * N_CLS;
  for (int i = tid; i < 4 * N_CLS; i += 128)
    out[dbase + (long)row * (4 * N_CLS) + i] = crow[N_CLS + i];     // deltas
}

// ---------------------------------------------------------------------------
extern "C" void kernel_launch(void* const* d_in, const int* in_sizes, int n_in,
                              void* d_out, int out_size, void* d_ws, size_t ws_size,
                              hipStream_t stream) {
  const float* pooled   = (const float*)d_in[0];   // [2000,7,7,256]
  const float* conv1_w  = (const float*)d_in[1];   // [12544,1024]
  const float* conv1_b  = (const float*)d_in[2];   // [1024]
  const float* bn1_g    = (const float*)d_in[3];
  const float* bn1_b    = (const float*)d_in[4];
  const float* conv2_w  = (const float*)d_in[5];   // [1024,1024]
  const float* conv2_b  = (const float*)d_in[6];
  const float* bn2_g    = (const float*)d_in[7];
  const float* bn2_b    = (const float*)d_in[8];
  const float* logits_w = (const float*)d_in[9];   // [1024,81]
  const float* logits_b = (const float*)d_in[10];
  const float* delta_w  = (const float*)d_in[11];  // [1024,324]
  const float* delta_b  = (const float*)d_in[12];
  float* out = (float*)d_out;

  // -------- workspace carve-up (all offsets multiple of 64 B) --------
  char* p = (char*)d_ws;
  _Float16* A1  = (_Float16*)p;  p += (size_t)M_ROWS * K_1 * 2;        // 50.2 MB
  _Float16* W1t = (_Float16*)p;  p += (size_t)H_DIM * K_1 * 2;         // 25.7 MB
  float*    C1  = (float*)p;     p += (size_t)M_ROWS * H_DIM * 4;      // 8.2 MB
  _Float16* Y1  = (_Float16*)p;  p += (size_t)M_ROWS * H_DIM * 2;      // 4.1 MB
  _Float16* W2t = (_Float16*)p;  p += (size_t)H_DIM * H_DIM * 2;       // 2.1 MB
  float*    C2  = (float*)p;     p += (size_t)M_ROWS * H_DIM * 4;      // 8.2 MB
  _Float16* Y2  = (_Float16*)p;  p += (size_t)M_ROWS * H_DIM * 2;      // 4.1 MB
  _Float16* W3t = (_Float16*)p;  p += (size_t)N_OUT3 * H_DIM * 2;      // 0.83 MB
  float*    C3  = (float*)p;     p += (size_t)M_ROWS * N_OUT3 * 4;     // 3.2 MB
  float*    bias3  = (float*)p;  p += 2048;
  float*    scale1 = (float*)p;  p += H_DIM * 4;
  float*    shift1 = (float*)p;  p += H_DIM * 4;
  float*    scale2 = (float*)p;  p += H_DIM * 4;
  float*    shift2 = (float*)p;  p += H_DIM * 4;

  // -------- stage operands as f16 (weights transposed to N x K) --------
  {
    long nA = (long)M_ROWS * K_1;
    cast_f16_kernel<<<4096, 256, 0, stream>>>(pooled, A1, nA);

    long n1 = (long)K_1 * H_DIM;
    tcast_kernel<<<(int)((n1 + 255) / 256), 256, 0, stream>>>(
        conv1_w, W1t, K_1, H_DIM, 0, K_1);

    long n2 = (long)H_DIM * H_DIM;
    tcast_kernel<<<(int)((n2 + 255) / 256), 256, 0, stream>>>(
        conv2_w, W2t, H_DIM, H_DIM, 0, H_DIM);

    long n3a = (long)H_DIM * N_CLS;
    tcast_kernel<<<(int)((n3a + 255) / 256), 256, 0, stream>>>(
        logits_w, W3t, H_DIM, N_CLS, 0, H_DIM);

    long n3b = (long)H_DIM * (4 * N_CLS);
    tcast_kernel<<<(int)((n3b + 255) / 256), 256, 0, stream>>>(
        delta_w, W3t, H_DIM, 4 * N_CLS, N_CLS, H_DIM);

    bias3_kernel<<<2, 256, 0, stream>>>(logits_b, delta_b, bias3);
  }

  // -------- GEMM1: [2000,12544] x [12544,1024] + conv1_b --------
  {
    dim3 grid((M_ROWS + 127) / 128, H_DIM / 128);
    wmma_gemm_kernel<<<grid, 256, 0, stream>>>(A1, W1t, conv1_b, C1,
                                               M_ROWS, H_DIM, K_1);
  }
  bn_stats_kernel<<<H_DIM, 256, 0, stream>>>(C1, bn1_g, bn1_b, scale1, shift1,
                                             M_ROWS, H_DIM);
  bn_relu_cast_kernel<<<2048, 256, 0, stream>>>(C1, scale1, shift1, Y1,
                                                H_DIM, (long)M_ROWS * H_DIM);

  // -------- GEMM2: [2000,1024] x [1024,1024] + conv2_b --------
  {
    dim3 grid((M_ROWS + 127) / 128, H_DIM / 128);
    wmma_gemm_kernel<<<grid, 256, 0, stream>>>(Y1, W2t, conv2_b, C2,
                                               M_ROWS, H_DIM, H_DIM);
  }
  bn_stats_kernel<<<H_DIM, 256, 0, stream>>>(C2, bn2_g, bn2_b, scale2, shift2,
                                             M_ROWS, H_DIM);
  bn_relu_cast_kernel<<<2048, 256, 0, stream>>>(C2, scale2, shift2, Y2,
                                                H_DIM, (long)M_ROWS * H_DIM);

  // -------- GEMM3 (fused logits|deltas): [2000,1024] x [1024,405] --------
  {
    dim3 grid((M_ROWS + 127) / 128, (N_OUT3 + 127) / 128);
    wmma_gemm_kernel<<<grid, 256, 0, stream>>>(Y2, W3t, bias3, C3,
                                               M_ROWS, N_OUT3, H_DIM);
  }

  // -------- softmax + scatter into concatenated d_out --------
  head_out_kernel<<<M_ROWS, 128, 0, stream>>>(C3, out);
}
